// TraceModel_82540681494887
// MI455X (gfx1250) — matrix operand order repaired and checked by reading, not compiled, not run on hardware
//
#include <hip/hip_runtime.h>

typedef __attribute__((ext_vector_type(16))) _Float16 v16h;
typedef __attribute__((ext_vector_type(8)))  float    v8f;

#define T_ 8
#define N_ 20000
#define E_ 320000
#define F_IN 128
#define F_HID 128
#define F_OUT 64
#define GRU_H 128
#define NCLS 16
#define ROWS (T_ * N_)   // 160000 total node-rows across snapshots

// ---------------------------------------------------------------- utilities
__global__ void fill_f32(float* __restrict__ p, float v, int n) {
    int i = blockIdx.x * blockDim.x + threadIdx.x;
    if (i < n) p[i] = v;
}

// degree: deg[t*N + dst] += 1  (deg pre-initialized to 1.0 for the self loop)
__global__ void count_deg(const int* __restrict__ ei, float* __restrict__ deg) {
    int i = blockIdx.x * blockDim.x + threadIdx.x;
    if (i >= T_ * E_) return;
    int t = i / E_, e = i - t * E_;
    int dst = ei[(size_t)t * 2 * E_ + E_ + e];
    atomicAdd(&deg[t * N_ + dst], 1.0f);
}

// ------------------------------------------------- pack W into WMMA B-layout
// B fragment (16-bit, 32x16): col n=lane&15; h=lane>=16: elems 0..7 -> K=8h+e,
// elems 8..15 -> K=16+8h+(e-8).  Wp[((ks*ntiles+nt)*32+lane)*16 + e]
__global__ void pack_w(const float* __restrict__ W, _Float16* __restrict__ Wp,
                       int N, int ntiles) {
    int idx = blockIdx.x * blockDim.x + threadIdx.x;
    int total = 4 * ntiles * 32 * 16;          // ksteps=4 (K=128)
    if (idx >= total) return;
    int e    = idx & 15;
    int lane = (idx >> 4) & 31;
    int nt   = (idx >> 9) % ntiles;
    int ks   = idx / (512 * ntiles);
    int k = ks * 32 + ((lane >= 16) ? 8 : 0) + ((e < 8) ? e : e + 8);
    int n = nt * 16 + (lane & 15);
    Wp[idx] = (_Float16)W[k * N + n];
}

// ------------------------------------------------------------ WMMA GEMM
// C[M x 16*NTILES] = (RELU?)A[M x 128] * W[128 x 16*NTILES]
// One 16-row M-tile per block; each wave owns TWO 16-col N-tiles so every
// loaded A fragment feeds two back-to-back v_wmma_f32_16x16x32_f16.
// blockDim.x = 32 * (NTILES/2).  K=128 -> 4 K-steps -> 8 WMMAs per wave.
template <bool RELU>
__global__ void gemm_wmma(const float* __restrict__ A,
                          const _Float16* __restrict__ Wp,
                          float* __restrict__ C, int N, int ntiles) {
    const int K = 128;
    int lane = threadIdx.x & 31;
    int nt0  = (threadIdx.x >> 5) * 2;     // first of this wave's two N-tiles
    int m0   = blockIdx.x * 16;
    int row  = m0 + (lane & 15);
    int kb   = (lane >= 16) ? 8 : 0;       // A-fragment K base per lane half
    const float* arow = A + (size_t)row * K;

    v8f acc0 = {}, acc1 = {};
#pragma unroll
    for (int ks = 0; ks < 4; ++ks) {
        int k0 = ks * 32 + kb;
        // A fragment: e=0..7 -> k0+e ; e=8..15 -> k0+16+(e-8)
        float4 lo0 = *(const float4*)(arow + k0);
        float4 lo1 = *(const float4*)(arow + k0 + 4);
        float4 hi0 = *(const float4*)(arow + k0 + 16);
        float4 hi1 = *(const float4*)(arow + k0 + 20);
        if (RELU) {
            lo0.x=fmaxf(lo0.x,0.f); lo0.y=fmaxf(lo0.y,0.f); lo0.z=fmaxf(lo0.z,0.f); lo0.w=fmaxf(lo0.w,0.f);
            lo1.x=fmaxf(lo1.x,0.f); lo1.y=fmaxf(lo1.y,0.f); lo1.z=fmaxf(lo1.z,0.f); lo1.w=fmaxf(lo1.w,0.f);
            hi0.x=fmaxf(hi0.x,0.f); hi0.y=fmaxf(hi0.y,0.f); hi0.z=fmaxf(hi0.z,0.f); hi0.w=fmaxf(hi0.w,0.f);
            hi1.x=fmaxf(hi1.x,0.f); hi1.y=fmaxf(hi1.y,0.f); hi1.z=fmaxf(hi1.z,0.f); hi1.w=fmaxf(hi1.w,0.f);
        }
        v16h a;
        a[0]=(_Float16)lo0.x;  a[1]=(_Float16)lo0.y;  a[2]=(_Float16)lo0.z;  a[3]=(_Float16)lo0.w;
        a[4]=(_Float16)lo1.x;  a[5]=(_Float16)lo1.y;  a[6]=(_Float16)lo1.z;  a[7]=(_Float16)lo1.w;
        a[8]=(_Float16)hi0.x;  a[9]=(_Float16)hi0.y;  a[10]=(_Float16)hi0.z; a[11]=(_Float16)hi0.w;
        a[12]=(_Float16)hi1.x; a[13]=(_Float16)hi1.y; a[14]=(_Float16)hi1.z; a[15]=(_Float16)hi1.w;

        const _Float16* wbase = Wp + ((size_t)(ks * ntiles + nt0) * 32 + lane) * 16;
        v16h b0 = *(const v16h*)(wbase);
        v16h b1 = *(const v16h*)(wbase + 32 * 16);   // next N-tile, same K-step

        acc0 = __builtin_amdgcn_wmma_f32_16x16x32_f16(
            false, a, false, b0, (short)0, acc0, false, false);
        acc1 = __builtin_amdgcn_wmma_f32_16x16x32_f16(
            false, a, false, b1, (short)0, acc1, false, false);
    }
    // D layout: VGPR r -> M = m0 + r + (lane>=16?8:0), N = nt*16 + (lane&15)
    int col   = nt0 * 16 + (lane & 15);
    int rbase = m0 + ((lane >= 16) ? 8 : 0);
#pragma unroll
    for (int r = 0; r < 8; ++r) {
        float* crow = C + (size_t)(rbase + r) * N + col;
        crow[0]  = acc0[r];
        crow[16] = acc1[r];
    }
}

// ---------------------------------------------- self-loop + bias init
// out = h / deg + b   (bias folded here; ReLU deferred to the consumer)
__global__ void init_out(const float* __restrict__ h, const float* __restrict__ deg,
                         const float* __restrict__ bias, float* __restrict__ out, int F) {
    size_t idx = (size_t)blockIdx.x * blockDim.x + threadIdx.x;
    if (idx >= (size_t)ROWS * F) return;
    size_t row = idx / F;
    int f = (int)(idx - row * F);
    out[idx] = h[idx] / deg[row] + bias[f];
}

// ------------------------------------------- edge gather + atomic scatter
// one thread per (t, e, 4-float chunk): out[dst] += h[src] * dinv(src)*dinv(dst)
__global__ void edge_agg(const float* __restrict__ h, const int* __restrict__ ei,
                         const float* __restrict__ deg, float* __restrict__ out, int F) {
    size_t idx = (size_t)blockIdx.x * blockDim.x + threadIdx.x;
    int F4 = F >> 2;
    if (idx >= (size_t)T_ * E_ * F4) return;
    size_t te = idx / F4;
    int f4 = (int)(idx - te * F4);
    int t = (int)(te / E_);
    int e = (int)(te - (size_t)t * E_);
    int src = ei[(size_t)t * 2 * E_ + e];
    int dst = ei[(size_t)t * 2 * E_ + E_ + e];
    float coef = rsqrtf(deg[t * N_ + src]) * rsqrtf(deg[t * N_ + dst]);
    float4 v = *(const float4*)(h + ((size_t)(t * N_ + src)) * F + f4 * 4);
    float* o = out + ((size_t)(t * N_ + dst)) * F + f4 * 4;
    atomicAdd(o + 0, v.x * coef);
    atomicAdd(o + 1, v.y * coef);
    atomicAdd(o + 2, v.z * coef);
    atomicAdd(o + 3, v.w * coef);
}

// --------------------------------------------- mean pool (relu folded)
// grid = T*40 blocks, each block sums 500 nodes for all 64 features.
__global__ void pool_mean(const float* __restrict__ out2, float* __restrict__ embs) {
    int t = blockIdx.x / 40, chunk = blockIdx.x % 40;
    int f = threadIdx.x & 63, sub = threadIdx.x >> 6;   // 4 node-lanes per f
    float s = 0.f;
#pragma unroll 1
    for (int i = 0; i < 125; ++i) {
        int n = chunk * 500 + i * 4 + sub;
        s += fmaxf(out2[((size_t)(t * N_ + n)) * F_OUT + f], 0.f);
    }
    __shared__ float red[256];
    red[threadIdx.x] = s;
    __syncthreads();
    if (sub == 0)
        atomicAdd(&embs[t * F_OUT + f], red[f] + red[64 + f] + red[128 + f] + red[192 + f]);
}

// ---------------------------------------------- GRU (T=8) + classifier
__global__ void gru_head(const float* __restrict__ embs,
                         const float* __restrict__ Wih, const float* __restrict__ Whh,
                         const float* __restrict__ bih, const float* __restrict__ bhh,
                         const float* __restrict__ Wc,  const float* __restrict__ bc,
                         float* __restrict__ out) {
    __shared__ float h[GRU_H], x[F_OUT], gi[3 * GRU_H], gh[3 * GRU_H], hn[GRU_H];
    int tid = threadIdx.x;                       // 384 threads
    if (tid < GRU_H) h[tid] = 0.f;
    for (int t = 0; t < T_; ++t) {
        __syncthreads();
        if (tid < F_OUT) x[tid] = embs[t * F_OUT + tid] * (1.0f / (float)N_);
        __syncthreads();
        {
            float a = bih[tid];
            for (int k = 0; k < F_OUT; ++k) a += Wih[tid * F_OUT + k] * x[k];
            gi[tid] = a;
            float b = bhh[tid];
            for (int k = 0; k < GRU_H; ++k) b += Whh[tid * GRU_H + k] * h[k];
            gh[tid] = b;
        }
        __syncthreads();
        if (tid < GRU_H) {
            float r  = 1.f / (1.f + expf(-(gi[tid] + gh[tid])));
            float z  = 1.f / (1.f + expf(-(gi[GRU_H + tid] + gh[GRU_H + tid])));
            float nc = tanhf(gi[2 * GRU_H + tid] + r * gh[2 * GRU_H + tid]);
            hn[tid] = (1.f - z) * nc + z * h[tid];
        }
        __syncthreads();
        if (tid < GRU_H) h[tid] = hn[tid];
    }
    __syncthreads();
    if (tid < NCLS) {
        float a = bc[tid];
        for (int k = 0; k < GRU_H; ++k) a += Wc[tid * GRU_H + k] * h[k];
        out[tid] = a;
    }
}

// ---------------------------------------------------------------- driver
static inline size_t alignup(size_t v) { return (v + 255) & ~(size_t)255; }

extern "C" void kernel_launch(void* const* d_in, const int* in_sizes, int n_in,
                              void* d_out, int out_size, void* d_ws, size_t ws_size,
                              hipStream_t stream) {
    const float* x    = (const float*)d_in[0];
    const int*   ei   = (const int*)  d_in[1];
    const float* W1   = (const float*)d_in[2];
    const float* b1   = (const float*)d_in[3];
    const float* W2   = (const float*)d_in[4];
    const float* b2   = (const float*)d_in[5];
    const float* Wih  = (const float*)d_in[6];
    const float* Whh  = (const float*)d_in[7];
    const float* bih  = (const float*)d_in[8];
    const float* bhh  = (const float*)d_in[9];
    const float* Wc   = (const float*)d_in[10];
    const float* bc   = (const float*)d_in[11];
    float* out = (float*)d_out;

    // workspace carve-up
    char* ws = (char*)d_ws;
    size_t off = 0;
    float* deg  = (float*)(ws + off); off += alignup((size_t)ROWS * 4);
    float* hbuf = (float*)(ws + off); off += alignup((size_t)ROWS * F_HID * 4);  // h1, then h2
    float* obuf = (float*)(ws + off); off += alignup((size_t)ROWS * F_HID * 4);  // out1, then out2
    _Float16* wp1 = (_Float16*)(ws + off); off += alignup((size_t)4 * 8 * 32 * 16 * 2);
    _Float16* wp2 = (_Float16*)(ws + off); off += alignup((size_t)4 * 4 * 32 * 16 * 2);
    float* embs = (float*)(ws + off); off += alignup((size_t)T_ * F_OUT * 4);

    const int B = 256;
    // deg = 1 (self loop), embs = 0
    fill_f32<<<(ROWS + B - 1) / B, B, 0, stream>>>(deg, 1.0f, ROWS);
    fill_f32<<<(T_ * F_OUT + B - 1) / B, B, 0, stream>>>(embs, 0.0f, T_ * F_OUT);
    // pack weights into WMMA B-fragment layout
    pack_w<<<(4 * 8 * 512 + B - 1) / B, B, 0, stream>>>(W1, wp1, F_HID, 8);
    pack_w<<<(4 * 4 * 512 + B - 1) / B, B, 0, stream>>>(W2, wp2, F_OUT, 4);
    // degrees
    count_deg<<<(T_ * E_ + B - 1) / B, B, 0, stream>>>(ei, deg);

    // ---- conv1: h1 = x @ W1 ; out1 = scatter + self + b1  (relu deferred)
    gemm_wmma<false><<<ROWS / 16, 4 * 32, 0, stream>>>(x, wp1, hbuf, F_HID, 8);
    init_out<<<(size_t)ROWS * F_HID / B, B, 0, stream>>>(hbuf, deg, b1, obuf, F_HID);
    edge_agg<<<(size_t)T_ * E_ * (F_HID / 4) / B, B, 0, stream>>>(hbuf, ei, deg, obuf, F_HID);

    // ---- conv2: h2 = relu(out1) @ W2 ; out2 = scatter + self + b2
    gemm_wmma<true><<<ROWS / 16, 2 * 32, 0, stream>>>(obuf, wp2, hbuf, F_OUT, 4);
    init_out<<<(size_t)ROWS * F_OUT / B, B, 0, stream>>>(hbuf, deg, b2, obuf, F_OUT);
    edge_agg<<<(size_t)T_ * E_ * (F_OUT / 4) / B, B, 0, stream>>>(hbuf, ei, deg, obuf, F_OUT);

    // ---- mean pool (relu folded) + GRU + classifier
    pool_mean<<<T_ * 40, 256, 0, stream>>>(obuf, embs);
    gru_head<<<1, 3 * GRU_H, 0, stream>>>(embs, Wih, Whh, bih, bhh, Wc, bc, out);
}
